// Seq2Seq_62096637166356
// MI455X (gfx1250) — compile-verified, matrix operand order checked by reading, not run on hardware
//
#include <hip/hip_runtime.h>

// ---------------- model dims ----------------
// V=30000 L=400 B=16 T=145 E=100 EH=150 DH=200 ENC=300 COMB=500, SOS=1

typedef __attribute__((ext_vector_type(16))) __bf16 v16bf;
typedef __attribute__((ext_vector_type(8)))  float  v8f;

union BV { v16bf v; unsigned u[8]; };

// ---- gfx1250 async global->LDS path (guarded; falls back to direct loads) ----
#if defined(__gfx1250__) && __has_builtin(__builtin_amdgcn_global_load_async_to_lds_b128)
#define USE_ASYNC_LDS 1
#else
#define USE_ASYNC_LDS 0
#endif

#if __has_builtin(__builtin_amdgcn_s_wait_asynccnt)
#define ASYNC_WAIT(n) __builtin_amdgcn_s_wait_asynccnt(n)
#else
#define ASYNC_WAIT(n) asm volatile("s_wait_asynccnt %0" :: "i"(n) : "memory")
#endif

#if USE_ASYNC_LDS
typedef int v4i __attribute__((__vector_size__(16)));
typedef __attribute__((address_space(1))) v4i* gptr4;   // global int4*
typedef __attribute__((address_space(3))) v4i* lptr4;   // LDS int4*
__device__ __forceinline__ void async_cp16(const unsigned* gsrc, unsigned* lds){
  __builtin_amdgcn_global_load_async_to_lds_b128((gptr4)gsrc, (lptr4)lds, 0, 0);
}
#endif

__device__ __forceinline__ unsigned short f2bf(float f){
  unsigned u = __float_as_uint(f);
  unsigned r = u + 0x7FFFu + ((u >> 16) & 1u);
  return (unsigned short)(r >> 16);
}
__device__ __forceinline__ float bf2f(unsigned short h){
  return __uint_as_float(((unsigned)h) << 16);
}
// order-preserving float->uint key (handles negatives)
__device__ __forceinline__ unsigned fkey(float f){
  unsigned u = __float_as_uint(f);
  return (u & 0x80000000u) ? ~u : (u | 0x80000000u);
}
__device__ __forceinline__ float fdec(unsigned k){
  unsigned u = (k & 0x80000000u) ? (k & 0x7FFFFFFFu) : ~k;
  return __uint_as_float(u);
}
__device__ __forceinline__ float sigf(float x){ return 1.f/(1.f+expf(-x)); }

// per the ISA 16-bit A/B WMMA layout: 32-bit slot v of lane holds K pair starting at:
__device__ __forceinline__ int kmap(int lane, int v){
  return ((v >> 2) << 4) + ((lane >> 4) << 3) + ((v & 3) << 1);
}

// A: row-major bf16 [16][lda] (lda multiple of 32); gather per-lane pairs.
// (slots v=0..3 and v=4..7 are each 4 consecutive dwords -> compiler emits 2x b128)
__device__ __forceinline__ v16bf load_a(const unsigned short* A, int lda, int g){
  int lane = threadIdx.x & 31; int m = lane & 15;
  BV t;
#pragma unroll
  for (int v = 0; v < 8; ++v){
    int k = g*32 + kmap(lane, v);
    t.u[v] = *(const unsigned*)(A + m*lda + k);   // 4B-aligned: lda even, k even
  }
  return t.v;
}
// B panel pre-packed so each lane reads 8 consecutive dwords (32B).
__device__ __forceinline__ v16bf load_b(const unsigned* P, int nt, int g, int KG){
  const unsigned* p = P + ((size_t)(nt*KG + g)*32 + (threadIdx.x & 31))*8;
  BV t;
#pragma unroll
  for (int i = 0; i < 8; ++i) t.u[i] = p[i];
  return t.v;
}
__device__ __forceinline__ v8f wmma_bf16(v16bf a, v16bf b, v8f c){
  return __builtin_amdgcn_wmma_f32_16x16x32_bf16(false, a, false, b, (short)0, c, false, false);
}

// ---------------- weight panel packing ----------------
// B[k][n]: mode 0 -> W stored [N,K] (use W[n,k]); mode 1 -> W stored [K,N] (use W[k,n]).
// grid = NT*KG blocks, 256 threads; word (nt,g,lane,v) at dst[bid*256 + lane*8 + v].
__global__ void pack_b_panel(const float* __restrict__ W, unsigned* __restrict__ dst,
                             int N, int K, int KG, int mode){
  int bid = blockIdx.x; int nt = bid / KG, g = bid % KG;
  int tid = threadIdx.x; int lane = tid >> 3, v = tid & 7;
  int n  = nt*16 + (lane & 15);
  int k0 = g*32 + kmap(lane, v);
  float f0 = 0.f, f1 = 0.f;
  if (n < N){
    if (k0     < K) f0 = mode ? W[(size_t)k0*N + n]       : W[(size_t)n*K + k0];
    if (k0 + 1 < K) f1 = mode ? W[(size_t)(k0+1)*N + n]   : W[(size_t)n*K + k0 + 1];
  }
  dst[(size_t)bid*256 + tid] = (unsigned)f2bf(f0) | ((unsigned)f2bf(f1) << 16);
}

// ---------------- generic tiled GEMM: C[MT*16, NT*16] = A * panel (+bias) ----------------
__global__ void gemm_bf16(const unsigned short* __restrict__ Ab, int lda,
                          const unsigned* __restrict__ panel,
                          float* __restrict__ C, int ldc,
                          const float* __restrict__ bias,
                          int MT, int NT, int KG, int Nvalid){
  int w = (blockIdx.x*blockDim.x + threadIdx.x) >> 5;
  if (w >= MT*NT) return;                 // whole-wave uniform exit (EXEC stays full)
  int mt = w / NT, nt = w % NT;
  int lane = threadIdx.x & 31;
  const unsigned short* A = Ab + (size_t)mt*16*lda;
  v8f c = {};
  for (int g = 0; g < KG; ++g)
    c = wmma_bf16(load_a(A, lda, g), load_b(panel, nt, g, KG), c);
  int n = nt*16 + (lane & 15);
  float bv = (bias && n < Nvalid) ? bias[n] : 0.f;
  float* Cb = C + (size_t)mt*16*ldc;
  int mh = (lane >> 4)*8;
#pragma unroll
  for (int r = 0; r < 8; ++r) Cb[(size_t)(mh + r)*ldc + n] = c[r] + bv;
}

// ---------------- encoder ----------------
// embA[l][b][128] bf16 (padded A tiles, one per l since B==16)
__global__ void embed_kernel(const int* __restrict__ input, const float* __restrict__ emb,
                             unsigned short* __restrict__ embA){
  int idx = blockIdx.x*256 + threadIdx.x;
  if (idx >= 400*16*128) return;
  int k = idx & 127; int lb = idx >> 7; int b = lb & 15; int l = lb >> 4;
  int tok = input[l*16 + b];
  embA[idx] = (k < 100) ? f2bf(emb[(size_t)tok*100 + k]) : (unsigned short)0;
}

// one block per direction; recurrence with h held in LDS (packed-A bf16 + f32)
__global__ void __launch_bounds__(512) enc_gru_kernel(
    const unsigned* __restrict__ pWhhF, const unsigned* __restrict__ pWhhB,
    const float* __restrict__ giF, const float* __restrict__ giB,
    const float* __restrict__ bhhF, const float* __restrict__ bhhB,
    float* __restrict__ encOut, unsigned short* __restrict__ encHidA){
  __shared__ unsigned short hA[16*160];
  __shared__ float gh[16*464];
  __shared__ float hsh[16*150];
  const int dir = blockIdx.x;
  const unsigned* P  = dir ? pWhhB : pWhhF;
  const float*   gi  = dir ? giB   : giF;
  const float*   bhh = dir ? bhhB  : bhhF;
  int tid = threadIdx.x, lane = tid & 31, w = tid >> 5;
  for (int i = tid; i < 16*160; i += 512) hA[i] = 0;
  for (int i = tid; i < 16*150; i += 512) hsh[i] = 0.f;
  __syncthreads();
  for (int s = 0; s < 400; ++s){
    for (int nt = w; nt < 29; nt += 16){
      v8f c = {};
      for (int g = 0; g < 5; ++g)
        c = wmma_bf16(load_a((const unsigned short*)hA, 160, g), load_b(P, nt, g, 5), c);
      int n = nt*16 + (lane & 15); int mh = (lane >> 4)*8;
#pragma unroll
      for (int r = 0; r < 8; ++r) gh[(mh + r)*464 + n] = c[r];
    }
    __syncthreads();
    int gil = dir ? (399 - s) : s;
    for (int idx = tid; idx < 2400; idx += 512){
      int b = idx/150, j = idx%150;
      const float* g0 = gi + (size_t)(gil*16 + b)*464;    // bih already folded in
      float ghr = gh[b*464 + j]       + bhh[j];
      float ghz = gh[b*464 + 150 + j] + bhh[150 + j];
      float ghn = gh[b*464 + 300 + j] + bhh[300 + j];
      float r = sigf(g0[j] + ghr);
      float z = sigf(g0[150 + j] + ghz);
      float nn = tanhf(g0[300 + j] + r*ghn);
      float h = (1.f - z)*nn + z*hsh[b*150 + j];
      hsh[b*150 + j] = h;
      hA[b*160 + j]  = f2bf(h);
      encOut[(size_t)(gil*16 + b)*304 + dir*150 + j] = h;
    }
    __syncthreads();
  }
  for (int idx = tid; idx < 2400; idx += 512){
    int b = idx/150, j = idx%150;
    encHidA[b*320 + dir*150 + j] = f2bf(hsh[b*150 + j]);
  }
  if (dir == 0)
    for (int idx = tid; idx < 16*20; idx += 512)
      encHidA[(idx/20)*320 + 300 + (idx%20)] = 0;
}

__global__ void adapter_post(const float* __restrict__ h0C, float* __restrict__ hdecF,
                             unsigned short* __restrict__ hAdec, float* __restrict__ cover){
  for (int idx = threadIdx.x; idx < 16*224; idx += 256){
    int b = idx/224, j = idx%224;
    float v = (j < 200) ? h0C[b*208 + j] : 0.f;
    hdecF[idx] = v; hAdec[idx] = f2bf(v);
  }
  for (int idx = threadIdx.x; idx < 16*400; idx += 256) cover[idx] = 0.f;
}

// ---------------- decoder per-step kernels ----------------
__global__ void dec_embed(const float* __restrict__ emb, unsigned short* __restrict__ xA,
                          unsigned* __restrict__ maxred, float* __restrict__ sumred,
                          unsigned long long* __restrict__ argred, int step){
  __shared__ int toks[16];
  int tid = threadIdx.x;
  if (tid < 16)
    toks[tid] = (step == 0) ? 1
              : (int)(0xFFFFFFFFu - (unsigned)(argred[tid] & 0xFFFFFFFFull));
  __syncthreads();
  if (tid < 16){ maxred[tid] = 0u; sumred[tid] = 0.f; argred[tid] = 0ull; }
  for (int idx = tid; idx < 16*128; idx += 256){
    int b = idx >> 7, k = idx & 127;
    xA[idx] = (k < 100) ? f2bf(emb[(size_t)toks[b]*100 + k]) : (unsigned short)0;
  }
}

__global__ void dec_gates(const float* __restrict__ gi, const float* __restrict__ gh,
                          float* __restrict__ hdecF, unsigned short* __restrict__ hAdec){
  for (int idx = threadIdx.x; idx < 3200; idx += 256){
    int b = idx/200, j = idx%200;
    float r = sigf(gi[b*608 + j]       + gh[b*608 + j]);
    float z = sigf(gi[b*608 + 200 + j] + gh[b*608 + 200 + j]);
    float n = tanhf(gi[b*608 + 400 + j] + r*gh[b*608 + 400 + j]);
    float h = (1.f - z)*n + z*hdecF[b*224 + j];
    hdecF[b*224 + j] = h; hAdec[b*224 + j] = f2bf(h);
  }
}

__global__ void energy_kernel(const float* __restrict__ hW, const float* __restrict__ encOut,
                              const float* __restrict__ cover, const float* __restrict__ bilB,
                              const float* __restrict__ coverW, float* __restrict__ energyG){
  int idx = blockIdx.x*256 + threadIdx.x;
  if (idx >= 6400) return;
  int l = idx/16, b = idx%16;
  const float* eo = encOut + (size_t)(l*16 + b)*304;
  float s = 0.f;
  for (int e = 0; e < 300; ++e) s += hW[b*304 + e]*eo[e];
  energyG[l*16 + b] = s + bilB[0] + coverW[0]*logf(cover[b*400 + l] + 1e-31f);
}

__global__ void __launch_bounds__(512) attn_ctx_kernel(
    const float* __restrict__ energyG, const float* __restrict__ encOut,
    const float* __restrict__ hdecF, float* __restrict__ cover,
    float* __restrict__ attnG, float* __restrict__ pptrG,
    unsigned short* __restrict__ preA,
    const unsigned* __restrict__ pPre, const float* __restrict__ preB,
    const float* __restrict__ ptrW, const float* __restrict__ ptrB){
  __shared__ float buf[6400];              // attn [l][b]
  __shared__ unsigned short combA[16*512]; // [h | context | pad] bf16 packed A
  __shared__ float preL[16*112];
  int tid = threadIdx.x, lane = tid & 31, w = tid >> 5;
  { // wave w handles batch b=w : softmax over L, cover update
    int b = w;
    float m = -3.4e38f;
    for (int l = lane; l < 400; l += 32) m = fmaxf(m, energyG[l*16 + b]);
    m = fmaxf(m, __shfl_xor(m, 1, 32));  m = fmaxf(m, __shfl_xor(m, 2, 32));
    m = fmaxf(m, __shfl_xor(m, 4, 32));  m = fmaxf(m, __shfl_xor(m, 8, 32));
    m = fmaxf(m, __shfl_xor(m, 16, 32));
    float s = 0.f;
    for (int l = lane; l < 400; l += 32){
      float t = expf(energyG[l*16 + b] - m); buf[l*16 + b] = t; s += t;
    }
    s += __shfl_xor(s, 1, 32); s += __shfl_xor(s, 2, 32);
    s += __shfl_xor(s, 4, 32); s += __shfl_xor(s, 8, 32); s += __shfl_xor(s, 16, 32);
    float inv = 1.f/s;
    for (int l = lane; l < 400; l += 32){
      float a = buf[l*16 + b]*inv;
      buf[l*16 + b] = a; attnG[l*16 + b] = a; cover[b*400 + l] += a;
    }
  }
  __syncthreads();
  for (int idx = tid; idx < 4800; idx += 512){       // context
    int b = idx/300, e = idx%300;
    float acc = 0.f;
    for (int l = 0; l < 400; ++l) acc += buf[l*16 + b]*encOut[(size_t)(l*16 + b)*304 + e];
    combA[b*512 + 200 + e] = f2bf(acc);
  }
  for (int idx = tid; idx < 16*200; idx += 512){     // h part
    int b = idx/200, j = idx%200;
    combA[b*512 + j] = f2bf(hdecF[b*224 + j]);
  }
  for (int idx = tid; idx < 16*12; idx += 512)       // pad 500..511
    combA[(idx/12)*512 + 500 + (idx%12)] = 0;
  __syncthreads();
  if (w < 7){                                        // pre = combined @ preoutW^T
    v8f c = {};
    for (int g = 0; g < 16; ++g)
      c = wmma_bf16(load_a((const unsigned short*)combA, 512, g), load_b(pPre, w, g, 16), c);
    int n = w*16 + (lane & 15); int mh = (lane >> 4)*8;
    float bb = (n < 100) ? preB[n] : 0.f;
#pragma unroll
    for (int r = 0; r < 8; ++r) preL[(mh + r)*112 + n] = c[r] + bb;
  } else if (w == 7 && lane < 16){                   // p_ptr
    int b = lane; float acc = 0.f;
    for (int k = 0; k < 500; ++k) acc += bf2f(combA[b*512 + k])*ptrW[k];
    pptrG[b] = sigf(acc + ptrB[0]);
  }
  __syncthreads();
  for (int idx = tid; idx < 16*128; idx += 512){
    int b = idx >> 7, k = idx & 127;
    preA[idx] = (k < 100) ? f2bf(preL[b*112 + k]) : (unsigned short)0;
  }
}

// logits = pre @ Emb^T + out_b; B panels double-buffered through LDS with the
// gfx1250 async global->LDS copies (ASYNCcnt) so L2 latency hides under WMMA.
__global__ void __launch_bounds__(256) logits_kernel(
    const unsigned short* __restrict__ preA,
    const unsigned* __restrict__ pEmb,
    const float* __restrict__ outBias,
    float* __restrict__ buf, unsigned* __restrict__ maxred){
  int w    = (blockIdx.x*blockDim.x + threadIdx.x) >> 5;  // global n-tile
  int lw   = threadIdx.x >> 5;                            // wave in block
  int lane = threadIdx.x & 31;
  bool active = (w < 1875);
  __builtin_prefetch(preA, 0, 3);                         // global_prefetch_b8
  v16bf a[4];
#pragma unroll
  for (int g = 0; g < 4; ++g) a[g] = load_a(preA, 128, g); // A reused 4x (L2-hot)
  v8f c = {};
#if USE_ASYNC_LDS
  __shared__ unsigned stage[8*2*256];       // per wave: 2 x 1KB B-panel buffers
  unsigned* myb = stage + lw*2*256;
  if (active){
    const unsigned* s0 = pEmb + ((size_t)(w*4 + 0)*32 + lane)*8;
    async_cp16(s0,     myb + lane*8);
    async_cp16(s0 + 4, myb + lane*8 + 4);
#pragma unroll
    for (int g = 0; g < 4; ++g){
      if (g < 3){
        const unsigned* sn = pEmb + ((size_t)(w*4 + g + 1)*32 + lane)*8;
        unsigned* dn = myb + ((g + 1) & 1)*256 + lane*8;
        async_cp16(sn, dn);
        async_cp16(sn + 4, dn + 4);
        ASYNC_WAIT(2);                      // current buffer's 2 copies done
      } else {
        ASYNC_WAIT(0);
      }
      BV t;
      const unsigned* p = myb + (g & 1)*256 + lane*8;
#pragma unroll
      for (int i = 0; i < 8; ++i) t.u[i] = p[i];   // 2x ds_load_b128
      c = wmma_bf16(a[g], t.v, c);
    }
  }
#else
  if (active){
#pragma unroll
    for (int g = 0; g < 4; ++g)
      c = wmma_bf16(a[g], load_b(pEmb, w, g, 4), c);
  }
#endif
  if (!active) return;
  int n = w*16 + (lane & 15);
  float bb = outBias[n];
  int mh = (lane >> 4)*8;
#pragma unroll
  for (int r = 0; r < 8; ++r){
    float val = c[r] + bb;
    buf[(size_t)(mh + r)*30000 + n] = val;
    float mx = val;
    mx = fmaxf(mx, __shfl_xor(mx, 1, 32));
    mx = fmaxf(mx, __shfl_xor(mx, 2, 32));
    mx = fmaxf(mx, __shfl_xor(mx, 4, 32));
    mx = fmaxf(mx, __shfl_xor(mx, 8, 32));   // reduce over 16 lanes of this half
    if ((lane & 15) == 0) atomicMax(&maxred[mh + r], fkey(mx));
  }
}

__global__ void sumexp_kernel(float* __restrict__ buf, const unsigned* __restrict__ maxred,
                              float* __restrict__ sumred){
  int b = blockIdx.y; int v = blockIdx.x*256 + threadIdx.x;
  float e = 0.f;
  if (v < 30000){
    e = expf(buf[(size_t)b*30000 + v] - fdec(maxred[b]));
    buf[(size_t)b*30000 + v] = e;
  }
  __shared__ float red[256];
  red[threadIdx.x] = e; __syncthreads();
  for (int s = 128; s > 0; s >>= 1){
    if (threadIdx.x < s) red[threadIdx.x] += red[threadIdx.x + s];
    __syncthreads();
  }
  if (threadIdx.x == 0) atomicAdd(&sumred[b], red[0]);
}

__global__ void finalize_kernel(const float* __restrict__ buf, const float* __restrict__ sumred,
                                const float* __restrict__ pptr, float* __restrict__ outT){
  int b = blockIdx.y; int v = blockIdx.x*256 + threadIdx.x;
  if (v >= 30000) return;
  outT[(size_t)b*30000 + v] = (1.f - pptr[b]) * buf[(size_t)b*30000 + v] / sumred[b];
}

__global__ void scatter_kernel(float* __restrict__ outT, const int* __restrict__ input,
                               const float* __restrict__ attnG, const float* __restrict__ pptr){
  int idx = blockIdx.x*256 + threadIdx.x;
  if (idx >= 6400) return;
  int l = idx/16, b = idx%16;
  atomicAdd(&outT[(size_t)b*30000 + input[l*16 + b]], pptr[b]*attnG[l*16 + b]);
}

__global__ void argmax_kernel(const float* __restrict__ outT, unsigned long long* __restrict__ argred){
  int b = blockIdx.y; int v = blockIdx.x*256 + threadIdx.x;
  unsigned long long key = 0ull;
  if (v < 30000){
    float f = outT[(size_t)b*30000 + v];           // probabilities are >= 0
    key = ((unsigned long long)__float_as_uint(f) << 32)
        | (unsigned long long)(0xFFFFFFFFu - (unsigned)v); // first-max tie-break
  }
  __shared__ unsigned long long red[256];
  red[threadIdx.x] = key; __syncthreads();
  for (int s = 128; s > 0; s >>= 1){
    if (threadIdx.x < s){
      unsigned long long o = red[threadIdx.x + s];
      if (o > red[threadIdx.x]) red[threadIdx.x] = o;
    }
    __syncthreads();
  }
  if (threadIdx.x == 0) atomicMax(&argred[b], red[0]);
}

// ---------------- host launcher ----------------
extern "C" void kernel_launch(void* const* d_in, const int* in_sizes, int n_in,
                              void* d_out, int out_size, void* d_ws, size_t ws_size,
                              hipStream_t stream){
  const int*   input   = (const int*)  d_in[0];
  const float* emb     = (const float*)d_in[1];
  const float* eWihF   = (const float*)d_in[2];
  const float* eWhhF   = (const float*)d_in[3];
  const float* ebihF   = (const float*)d_in[4];
  const float* ebhhF   = (const float*)d_in[5];
  const float* eWihB   = (const float*)d_in[6];
  const float* eWhhB   = (const float*)d_in[7];
  const float* ebihB   = (const float*)d_in[8];
  const float* ebhhB   = (const float*)d_in[9];
  const float* adpW    = (const float*)d_in[10];
  const float* adpB    = (const float*)d_in[11];
  const float* dWih    = (const float*)d_in[12];
  const float* dWhh    = (const float*)d_in[13];
  const float* dbih    = (const float*)d_in[14];
  const float* dbhh    = (const float*)d_in[15];
  const float* bilW    = (const float*)d_in[16];
  const float* bilB    = (const float*)d_in[17];
  const float* coverW  = (const float*)d_in[18];
  const float* ptrW    = (const float*)d_in[19];
  const float* ptrB    = (const float*)d_in[20];
  const float* preW    = (const float*)d_in[21];
  const float* preB    = (const float*)d_in[22];
  const float* outBias = (const float*)d_in[23];
  float* dout = (float*)d_out;
  (void)in_sizes; (void)n_in; (void)out_size; (void)ws_size;

  char* base = (char*)d_ws; size_t off = 0;
  auto WS = [&](size_t n)->char*{ char* p = base + off; off = (off + n + 255) & ~(size_t)255; return p; };

  // packed bf16 B panels (dwords)
  unsigned* pEmb  = (unsigned*)WS((size_t)1875*4*256*4);   // Emb^T  K=100->128, N=30000
  unsigned* pWihF = (unsigned*)WS((size_t)29*4*256*4);     // enc Wih K=100, N=450
  unsigned* pWhhF = (unsigned*)WS((size_t)29*5*256*4);     // enc Whh K=150->160
  unsigned* pWihB = (unsigned*)WS((size_t)29*4*256*4);
  unsigned* pWhhB = (unsigned*)WS((size_t)29*5*256*4);
  unsigned* pDWih = (unsigned*)WS((size_t)38*4*256*4);     // dec Wih K=100, N=600
  unsigned* pDWhh = (unsigned*)WS((size_t)38*7*256*4);     // dec Whh K=200->224
  unsigned* pBil  = (unsigned*)WS((size_t)19*7*256*4);     // bilW [200,300] (mode 1)
  unsigned* pAdp  = (unsigned*)WS((size_t)13*10*256*4);    // adapter K=300->320, N=200
  unsigned* pPre  = (unsigned*)WS((size_t)7*16*256*4);     // preout K=500->512, N=100
  // activations
  unsigned short* embA    = (unsigned short*)WS((size_t)400*16*128*2);
  float*          giF     = (float*)WS((size_t)400*16*464*4);
  float*          giB     = (float*)WS((size_t)400*16*464*4);
  float*          encOut  = (float*)WS((size_t)400*16*304*4);
  unsigned short* encHidA = (unsigned short*)WS((size_t)16*320*2);
  float*          h0C     = (float*)WS((size_t)16*208*4);
  float*          hdecF   = (float*)WS((size_t)16*224*4);
  unsigned short* hAdec   = (unsigned short*)WS((size_t)16*224*2);
  unsigned short* xA      = (unsigned short*)WS((size_t)16*128*2);
  float*          giD     = (float*)WS((size_t)16*608*4);
  float*          ghD     = (float*)WS((size_t)16*608*4);
  float*          hWbuf   = (float*)WS((size_t)16*304*4);
  float*          energyG = (float*)WS((size_t)400*16*4);
  float*          attnG   = (float*)WS((size_t)400*16*4);
  float*          cover   = (float*)WS((size_t)16*400*4);
  unsigned short* preA    = (unsigned short*)WS((size_t)16*128*2);
  float*          pptrG   = (float*)WS((size_t)16*4);
  unsigned*       maxred  = (unsigned*)WS((size_t)16*4);
  float*          sumred  = (float*)WS((size_t)16*4);
  unsigned long long* argred = (unsigned long long*)WS((size_t)16*8);
  float*          logbuf  = (float*)WS((size_t)16*30000*4);

  auto pack = [&](const float* W, unsigned* dst, int N, int K, int NT, int KG, int mode){
    pack_b_panel<<<NT*KG, 256, 0, stream>>>(W, dst, N, K, KG, mode);
  };
  // -------- one-time (per launch) weight packing --------
  pack(emb,   pEmb,  30000, 100, 1875, 4, 0);
  pack(eWihF, pWihF, 450, 100, 29, 4, 0);
  pack(eWhhF, pWhhF, 450, 150, 29, 5, 0);
  pack(eWihB, pWihB, 450, 100, 29, 4, 0);
  pack(eWhhB, pWhhB, 450, 150, 29, 5, 0);
  pack(dWih,  pDWih, 600, 100, 38, 4, 0);
  pack(dWhh,  pDWhh, 600, 200, 38, 7, 0);
  pack(bilW,  pBil,  300, 200, 19, 7, 1);   // stored [K=200, N=300]
  pack(adpW,  pAdp,  200, 300, 13, 10, 0);
  pack(preW,  pPre,  100, 500, 7, 16, 0);

  // -------- encoder --------
  embed_kernel<<<3200, 256, 0, stream>>>(input, emb, embA);
  // gi for all 400 steps, both directions (parallel batched GEMMs)
  gemm_bf16<<<1450, 256, 0, stream>>>(embA, 128, pWihF, giF, 464, ebihF, 400, 29, 4, 450);
  gemm_bf16<<<1450, 256, 0, stream>>>(embA, 128, pWihB, giB, 464, ebihB, 400, 29, 4, 450);
  enc_gru_kernel<<<2, 512, 0, stream>>>(pWhhF, pWhhB, giF, giB, ebhhF, ebhhB, encOut, encHidA);
  gemm_bf16<<<2, 256, 0, stream>>>(encHidA, 320, pAdp, h0C, 208, adpB, 1, 13, 10, 200);
  adapter_post<<<1, 256, 0, stream>>>(h0C, hdecF, hAdec, cover);

  // -------- decoder: T sequential steps --------
  dim3 gv(118, 16);
  for (int t = 0; t < 145; ++t){
    dec_embed<<<1, 256, 0, stream>>>(emb, xA, maxred, sumred, argred, t);
    gemm_bf16<<<5, 256, 0, stream>>>(xA,    128, pDWih, giD, 608, dbih, 1, 38, 4, 600);
    gemm_bf16<<<5, 256, 0, stream>>>(hAdec, 224, pDWhh, ghD, 608, dbhh, 1, 38, 7, 600);
    dec_gates<<<1, 256, 0, stream>>>(giD, ghD, hdecF, hAdec);
    gemm_bf16<<<3, 256, 0, stream>>>(hAdec, 224, pBil, hWbuf, 304, nullptr, 1, 19, 7, 300);
    energy_kernel<<<25, 256, 0, stream>>>(hWbuf, encOut, cover, bilB, coverW, energyG);
    attn_ctx_kernel<<<1, 512, 0, stream>>>(energyG, encOut, hdecF, cover, attnG, pptrG,
                                           preA, pPre, preB, ptrW, ptrB);
    logits_kernel<<<235, 256, 0, stream>>>(preA, pEmb, outBias, logbuf, maxred);
    sumexp_kernel<<<gv, 256, 0, stream>>>(logbuf, maxred, sumred);
    float* outT = dout + (size_t)t*16*30000;
    finalize_kernel<<<gv, 256, 0, stream>>>(logbuf, sumred, pptrG, outT);
    scatter_kernel<<<25, 256, 0, stream>>>(outT, input, attnG, pptrG);
    argmax_kernel<<<gv, 256, 0, stream>>>(outT, argred);
  }
}
